// Generator_61555471286544
// MI455X (gfx1250) — matrix operand order repaired and checked by reading, not compiled, not run on hardware
//
#include <hip/hip_runtime.h>
#include <hip/hip_bf16.h>
#include <cstdint>

#define BB     131072
#define NLAYER 4
#define NCAT   8
#define LODD   4
#define HHDIM  64
#define VDD    16
#define DDIM   113
#define TAUF   0.1f

#define NTHREADS 512                 // 16 waves; 256 samples per block
#define NWAVES   (NTHREADS/32)
#define SPB      (NWAVES*16)         // samples per block

typedef __attribute__((ext_vector_type(16))) _Float16 v16h;
typedef __attribute__((ext_vector_type(8)))  _Float16 v8h;
typedef __attribute__((ext_vector_type(8)))  float    v8f;

// ---------------- LDS layout (offsets in _Float16 units) ----------------
// Weights stored TRANSPOSED [N][Kpad] (f16) so A and B fragments are
// contiguous 16B ds loads.
#define OFF_LW0   0        // [4][64][32]  (K=8  padded to 32)
#define OFF_LW1   8192     // [4][64][64]
#define OFF_LW2S  24576    // [4][16][64]  (4 selected cols, rest zero; selval folded)
#define OFF_NW0   28672    // [6][64][32]  (K=20 padded to 32)
#define OFF_NW1   40960    // [6][64][64]
#define OFF_NWF   65536    // [5][16][64]
#define OFF_W7F   70656    // [16][64]     (2 cols valid)
#define OFF_SCR   71680    // per-wave scratch, stride 3072:
                           //   tA 16x32 | tH 16x64 | tH2 16x64 | tV 16x16 | tL 4x16x4
#define OFF_F32   (OFF_SCR + NWAVES*3072)   // f32 region starts here (half-offset)
#define SMEM_BYTES (OFF_F32*2 + 1408*4)

union FragU { v16h v; v8h h[2]; };
union AccU  { v8f  v; float f[8]; };

// A fragment: lane holds row M=lane%16; K = kbase + hi*8 + {e, e+8 splits}
// -> two contiguous 8-half (16B) chunks at p and p+16.
__device__ __forceinline__ v16h fragA(const _Float16* p) {
  FragU u; u.h[0] = *(const v8h*)p; u.h[1] = *(const v8h*)(p + 16); return u.v;
}
// B fragment: lane holds col N=lane%16; K = kbase + hi*8 + {e, e+8 splits} (same as A).
__device__ __forceinline__ v16h fragB(const _Float16* p) {
  FragU u; u.h[0] = *(const v8h*)p; u.h[1] = *(const v8h*)(p + 16); return u.v;
}
__device__ __forceinline__ v8f wmma16(v16h a, v16h b, v8f c) {
  return __builtin_amdgcn_wmma_f32_16x16x32_f16(false, a, false, b, (short)0, c,
                                                false, false);
}

__global__ __launch_bounds__(NTHREADS, 1)
void scm_gen_kernel(const float* __restrict__ z,    const float* __restrict__ data,
                    const float* __restrict__ glog, const float* __restrict__ glat,
                    const float* __restrict__ gbin,
                    const float* __restrict__ LW0g, const float* __restrict__ Lb0g,
                    const float* __restrict__ LW1g, const float* __restrict__ Lb1g,
                    const float* __restrict__ LW2g, const float* __restrict__ Lb2g,
                    const float* __restrict__ NW0g, const float* __restrict__ Nb0g,
                    const float* __restrict__ NW1g, const float* __restrict__ Nb1g,
                    const float* __restrict__ NWfg, const float* __restrict__ Nbfg,
                    const float* __restrict__ W7fg, const float* __restrict__ b7fg,
                    float* __restrict__ out)
{
  extern __shared__ _Float16 sw[];
  float* sbias = (float*)(sw + OFF_F32);
  float* sLb0  = sbias;          // 256
  float* sLb1  = sbias + 256;    // 256
  float* sLb2S = sbias + 512;    // 16
  float* sNb0  = sbias + 528;    // 384
  float* sNb1  = sbias + 912;    // 384
  float* sNbf  = sbias + 1296;   // 80
  float* sb7f  = sbias + 1376;   // 2
  float* sSelV = sbias + 1378;   // 4
  int*   sSelI = (int*)(sbias + 1382); // 4

  const int tid = threadIdx.x;
  const int NT  = blockDim.x;

  // ---- Phase 0a: straight-through Gumbel selection per latent layer ----
  if (tid < NLAYER) {
    const int l = tid;
    float p[NCAT]; float mx = -3.4e38f; int ks = 0;
    #pragma unroll
    for (int k = 0; k < NCAT; ++k) {
      p[k] = glog[l*NCAT + k] + glat[l*NCAT + k];
      if (p[k] > mx) { mx = p[k]; ks = k; }
    }
    float es = 0.f, ek = 0.f;
    #pragma unroll
    for (int k = 0; k < NCAT; ++k) {
      float e = __expf((p[k] - mx) / TAUF);
      es += e; if (k == ks) ek = e;
    }
    float y = ek / es;
    sSelV[l] = (1.0f + y) - y;   // forward value of hard+y-stop_grad(y) at argmax
    sSelI[l] = ks;               // off-argmax entries are exactly 0
  }
  __syncthreads();

  // ---- Phase 0b: cooperative weight staging (f32 -> f16, transposed) ----
  for (int i = tid; i < NLAYER*64*32; i += NT) {          // LW0 [l][n][k32]
    int l = i >> 11, r = i & 2047, n = r >> 5, k = r & 31;
    sw[OFF_LW0 + i] = (k < 8) ? (_Float16)LW0g[l*512 + k*64 + n] : (_Float16)0.f;
  }
  for (int i = tid; i < NLAYER*64*64; i += NT) {          // LW1 [l][n][k64]
    int l = i >> 12, r = i & 4095, n = r >> 6, k = r & 63;
    sw[OFF_LW1 + i] = (_Float16)LW1g[l*4096 + k*64 + n];
  }
  for (int i = tid; i < NLAYER*16*64; i += NT) {          // LW2 selected cols
    int l = i >> 10, r = i & 1023, n = r >> 6, k = r & 63;
    float v = (n < 4) ? LW2g[l*2048 + k*32 + (sSelI[l]*4 + n)] * sSelV[l] : 0.f;
    sw[OFF_LW2S + i] = (_Float16)v;
  }
  for (int i = tid; i < 6*64*32; i += NT) {               // NW0 [j][n][k32]
    int j = i >> 11, r = i & 2047, n = r >> 5, k = r & 31;
    sw[OFF_NW0 + i] = (k < 20) ? (_Float16)NW0g[j*1280 + k*64 + n] : (_Float16)0.f;
  }
  for (int i = tid; i < 6*64*64; i += NT) {               // NW1 [j][n][k64]
    int j = i >> 12, r = i & 4095, n = r >> 6, k = r & 63;
    sw[OFF_NW1 + i] = (_Float16)NW1g[j*4096 + k*64 + n];
  }
  for (int i = tid; i < 5*16*64; i += NT) {               // NWf [j][n16][k64]
    int j = i >> 10, r = i & 1023, n = r >> 6, k = r & 63;
    sw[OFF_NWF + i] = (_Float16)NWfg[j*1024 + k*16 + n];
  }
  for (int i = tid; i < 16*64; i += NT) {                 // W7f [n16][k64]
    int n = i >> 6, k = i & 63;
    sw[OFF_W7F + i] = (n < 2) ? (_Float16)W7fg[k*2 + n] : (_Float16)0.f;
  }
  for (int i = tid; i < 256; i += NT) sLb0[i] = Lb0g[i];
  for (int i = tid; i < 256; i += NT) sLb1[i] = Lb1g[i];
  if (tid < 16) { int l = tid >> 2, o = tid & 3;
    sLb2S[tid] = Lb2g[l*32 + sSelI[l]*4 + o] * sSelV[l]; }
  for (int i = tid; i < 384; i += NT) sNb0[i] = Nb0g[i];
  for (int i = tid; i < 384; i += NT) sNb1[i] = Nb1g[i];
  for (int i = tid; i < 80;  i += NT) sNbf[i] = Nbfg[i];
  if (tid < 2) sb7f[tid] = b7fg[tid];
  __syncthreads();

  // ---- Phase 1: per-wave, 16 samples each, whole network ----
  const int wave = tid >> 5;
  const int lane = tid & 31;
  const int col  = lane & 15;          // C/D col, B col, A row
  const int hi   = lane >> 4;
  const int hi8  = hi * 8;             // C row offset & A K-offset
  const int hi16 = hi * 8;             // B K-offset
  const int sBase = blockIdx.x * SPB + wave * 16;

  _Float16* tA  = sw + OFF_SCR + wave * 3072;  // 16x32 input staging
  _Float16* tH  = tA + 512;                    // 16x64 hidden 1
  _Float16* tH2 = tA + 1536;                   // 16x64 hidden 2
  _Float16* tV  = tA + 2560;                   // 16x16 current observed var
  _Float16* tL  = tA + 2816;                   // [4][16][4] latent outputs w

  // var0, var1: straight copy of data[:, 0:32] to out[:, 0:32]
  for (int i = lane; i < 16*32; i += 32) {
    int r = i >> 5, c = i & 31;
    out[(size_t)(sBase + r)*DDIM + c] = data[(size_t)(sBase + r)*DDIM + c];
  }

  // ---- latent MLPs: per layer l, 16x8 @ (8x64) -> 64 -> selected 4 ----
  for (int l = 0; l < NLAYER; ++l) {
    for (int i = lane; i < 512; i += 32) {         // stage z tile (K pad 32)
      int r = i >> 5, c = i & 31;
      tA[i] = (c < 8) ? (_Float16)z[(size_t)(sBase + r)*32 + l*8 + c]
                      : (_Float16)0.f;
    }
    const _Float16* W0 = sw + OFF_LW0 + l*2048;
    {
      v16h a0 = fragA(tA + col*32 + hi8);
      #pragma unroll
      for (int nt = 0; nt < 4; ++nt) {
        AccU acc; acc.v = v8f{};
        acc.v = wmma16(a0, fragB(W0 + (nt*16 + col)*32 + hi16), acc.v);
        float bsv = sLb0[l*64 + nt*16 + col];
        #pragma unroll
        for (int r = 0; r < 8; ++r)
          tH[(r + hi8)*64 + nt*16 + col] = (_Float16)fmaxf(acc.f[r] + bsv, 0.f);
      }
    }
    const _Float16* W1 = sw + OFF_LW1 + l*4096;
    {
      v16h a0 = fragA(tH + col*64 + 0  + hi8);
      v16h a1 = fragA(tH + col*64 + 32 + hi8);
      #pragma unroll
      for (int nt = 0; nt < 4; ++nt) {
        AccU acc; acc.v = v8f{};
        acc.v = wmma16(a0, fragB(W1 + (nt*16 + col)*64 + 0  + hi16), acc.v);
        acc.v = wmma16(a1, fragB(W1 + (nt*16 + col)*64 + 32 + hi16), acc.v);
        float bsv = sLb1[l*64 + nt*16 + col];
        #pragma unroll
        for (int r = 0; r < 8; ++r)
          tH2[(r + hi8)*64 + nt*16 + col] = (_Float16)fmaxf(acc.f[r] + bsv, 0.f);
      }
    }
    const _Float16* W2 = sw + OFF_LW2S + l*1024;   // selected 4 cols (N tile 16)
    {
      v16h a0 = fragA(tH2 + col*64 + 0  + hi8);
      v16h a1 = fragA(tH2 + col*64 + 32 + hi8);
      AccU acc; acc.v = v8f{};
      acc.v = wmma16(a0, fragB(W2 + col*64 + 0  + hi16), acc.v);
      acc.v = wmma16(a1, fragB(W2 + col*64 + 32 + hi16), acc.v);
      if (col < 4) {
        float bsv = sLb2S[l*4 + col];
        #pragma unroll
        for (int r = 0; r < 8; ++r)
          tL[l*64 + (r + hi8)*4 + col] = (_Float16)(acc.f[r] + bsv);
      }
    }
  }

  // init tV with var0 = data[:, 0:16]
  for (int i = lane; i < 256; i += 32) {
    int r = i >> 4, c = i & 15;
    tV[i] = (_Float16)data[(size_t)(sBase + r)*DDIM + c];
  }

  // ---- node chain: nodes 2..7, each consumes previous var + latent ----
  const int PLAT[6] = {0, 1, 2, 3, 0, 1};          // PARENT_LAT - 1
  for (int j = 0; j < 6; ++j) {
    const int node = j + 2;
    const int latl = PLAT[j];
    for (int i = lane; i < 512; i += 32) {         // stage [lat(4)|var(16)|0..]
      int r = i >> 5, c = i & 31;
      _Float16 v;
      if (c < 4)       v = tL[latl*64 + r*4 + c];
      else if (c < 20) v = tV[r*16 + (c - 4)];
      else             v = (_Float16)0.f;
      tA[i] = v;
    }
    const _Float16* W0 = sw + OFF_NW0 + j*2048;
    {
      v16h a0 = fragA(tA + col*32 + hi8);
      #pragma unroll
      for (int nt = 0; nt < 4; ++nt) {
        AccU acc; acc.v = v8f{};
        acc.v = wmma16(a0, fragB(W0 + (nt*16 + col)*32 + hi16), acc.v);
        float bsv = sNb0[j*64 + nt*16 + col];
        #pragma unroll
        for (int r = 0; r < 8; ++r)
          tH[(r + hi8)*64 + nt*16 + col] = (_Float16)fmaxf(acc.f[r] + bsv, 0.f);
      }
    }
    const _Float16* W1 = sw + OFF_NW1 + j*4096;
    {
      v16h a0 = fragA(tH + col*64 + 0  + hi8);
      v16h a1 = fragA(tH + col*64 + 32 + hi8);
      #pragma unroll
      for (int nt = 0; nt < 4; ++nt) {
        AccU acc; acc.v = v8f{};
        acc.v = wmma16(a0, fragB(W1 + (nt*16 + col)*64 + 0  + hi16), acc.v);
        acc.v = wmma16(a1, fragB(W1 + (nt*16 + col)*64 + 32 + hi16), acc.v);
        float bsv = sNb1[j*64 + nt*16 + col];
        #pragma unroll
        for (int r = 0; r < 8; ++r)
          tH2[(r + hi8)*64 + nt*16 + col] = (_Float16)fmaxf(acc.f[r] + bsv, 0.f);
      }
    }
    v16h a0 = fragA(tH2 + col*64 + 0  + hi8);
    v16h a1 = fragA(tH2 + col*64 + 32 + hi8);
    if (node != 7) {
      const _Float16* WF = sw + OFF_NWF + j*1024;
      AccU acc; acc.v = v8f{};
      acc.v = wmma16(a0, fragB(WF + col*64 + 0  + hi16), acc.v);
      acc.v = wmma16(a1, fragB(WF + col*64 + 32 + hi16), acc.v);
      float bsv = sNbf[j*16 + col];
      #pragma unroll
      for (int r = 0; r < 8; ++r) {
        float v = acc.f[r] + bsv;
        out[(size_t)(sBase + r + hi8)*DDIM + node*16 + col] = v;
        tV[(r + hi8)*16 + col] = (_Float16)v;
      }
    } else {
      const _Float16* W7 = sw + OFF_W7F;
      AccU acc; acc.v = v8f{};
      acc.v = wmma16(a0, fragB(W7 + col*64 + 0  + hi16), acc.v);
      acc.v = wmma16(a1, fragB(W7 + col*64 + 32 + hi16), acc.v);
      float* lbuf = (float*)tH;                    // 16x2 f32 logit buffer
      if (col < 2) {
        #pragma unroll
        for (int r = 0; r < 8; ++r)
          lbuf[(r + hi8)*2 + col] = acc.f[r] + sb7f[col];
      }
      if (lane < 16) {                             // per-sample binary ST-gumbel
        const int s = sBase + lane;
        float l0 = lbuf[lane*2 + 0], l1 = lbuf[lane*2 + 1];
        float q0 = 1.f/(1.f + __expf(-l0)) + gbin[(size_t)s*2 + 0];
        float q1 = 1.f/(1.f + __expf(-l1)) + gbin[(size_t)s*2 + 1];
        float mx = fmaxf(q0, q1);
        float e0 = __expf((q0 - mx)/TAUF), e1 = __expf((q1 - mx)/TAUF);
        float y0 = e0 / (e0 + e1);
        float v7 = (q1 > q0) ? 0.f : ((1.f + y0) - y0);  // argmax first-max tiebreak
        out[(size_t)s*DDIM + 112] = v7;
      }
    }
  }
}

extern "C" void kernel_launch(void* const* d_in, const int* in_sizes, int n_in,
                              void* d_out, int out_size, void* d_ws, size_t ws_size,
                              hipStream_t stream) {
  const float* z    = (const float*)d_in[0];
  const float* data = (const float*)d_in[1];
  const float* glog = (const float*)d_in[2];
  const float* glat = (const float*)d_in[3];
  const float* gbin = (const float*)d_in[4];
  const float* LW0  = (const float*)d_in[5];
  const float* Lb0  = (const float*)d_in[6];
  const float* LW1  = (const float*)d_in[7];
  const float* Lb1  = (const float*)d_in[8];
  const float* LW2  = (const float*)d_in[9];
  const float* Lb2  = (const float*)d_in[10];
  const float* NW0  = (const float*)d_in[11];
  const float* Nb0  = (const float*)d_in[12];
  const float* NW1  = (const float*)d_in[13];
  const float* Nb1  = (const float*)d_in[14];
  const float* NWf  = (const float*)d_in[15];
  const float* Nbf  = (const float*)d_in[16];
  const float* W7f  = (const float*)d_in[17];
  const float* b7f  = (const float*)d_in[18];
  float* out = (float*)d_out;

  (void)hipFuncSetAttribute((const void*)scm_gen_kernel,
                            hipFuncAttributeMaxDynamicSharedMemorySize,
                            SMEM_BYTES);
  scm_gen_kernel<<<BB/SPB, NTHREADS, SMEM_BYTES, stream>>>(
      z, data, glog, glat, gbin, LW0, Lb0, LW1, Lb1, LW2, Lb2,
      NW0, Nb0, NW1, Nb1, NWf, Nbf, W7f, b7f, out);
}
